// HyperComplexAdapterBlock_17188459119171
// MI455X (gfx1250) — matrix operand order repaired and checked by reading, not compiled, not run on hardware
//
#include <hip/hip_runtime.h>
#include <hip/hip_bf16.h>

// MI455X (gfx1250) fused PHM adapter block.
// out = gelu_new(x @ H_down + b_d) @ H_up + b_u,  H_* precomputed (rank-1 PHM collapse).
// x:[32768,768] f32, H_down:[768,192], H_up:[192,768] stored transposed bf16 in ws (L2-resident).
// Each wave owns 32 rows (2 A-fragments) so every LDS B-fragment feeds 2 WMMAs.

typedef __attribute__((ext_vector_type(16))) __bf16       v16bf;
typedef __attribute__((ext_vector_type(2)))  __bf16       v2bf;
typedef __attribute__((ext_vector_type(8)))  float        v8f;
typedef __attribute__((ext_vector_type(4)))  float        f32x4;
typedef __attribute__((ext_vector_type(4)))  unsigned int u32x4;
typedef __attribute__((ext_vector_type(4)))  int          i32x4;

typedef __attribute__((address_space(1))) i32x4 gi32x4;   // global int4
typedef __attribute__((address_space(3))) i32x4 si32x4;   // LDS int4

#if defined(__has_builtin)
#if __has_builtin(__builtin_amdgcn_cvt_pk_bf16_f32)
#define HAVE_CVT_PK_BF16 1
#endif
#if __has_builtin(__builtin_amdgcn_global_load_async_to_lds_b128)
#define HAVE_ASYNC_LDS 1
#endif
#endif
#ifndef HAVE_CVT_PK_BF16
#define HAVE_CVT_PK_BF16 0
#endif
#ifndef HAVE_ASYNC_LDS
#define HAVE_ASYNC_LDS 0
#endif

union Frag { unsigned int u[8]; v16bf v; };

__device__ __forceinline__ unsigned short f2bf(float f) {
    unsigned int u = __float_as_uint(f);
    return (unsigned short)((u + 0x8000u) >> 16);   // round-to-nearest (ties up)
}
__device__ __forceinline__ unsigned int pack2(float lo, float hi) {
#if HAVE_CVT_PK_BF16
    v2bf p = __builtin_amdgcn_cvt_pk_bf16_f32(lo, hi);
    return __builtin_bit_cast(unsigned int, p);
#else
    unsigned int l = (__float_as_uint(lo) + 0x8000u) >> 16;
    unsigned int h = (__float_as_uint(hi) + 0x8000u) & 0xFFFF0000u;
    return h | l;
#endif
}
__device__ __forceinline__ float gelu_new_f(float v) {
    float t  = 0.7978845608028654f * fmaf(0.044715f * v * v, v, v);
    float e  = __expf(2.0f * t);                              // v_exp_f32
    float th = 1.0f - 2.0f * __builtin_amdgcn_rcpf(e + 1.0f); // v_rcp_f32
    return 0.5f * v * (1.0f + th);
}

// 16B global -> LDS: async DMA (ASYNCcnt), overlapped with WMMA on the other buffer.
__device__ __forceinline__ void ld_b128_to_lds(const unsigned int* g, unsigned int* l) {
#if HAVE_ASYNC_LDS
    __builtin_amdgcn_global_load_async_to_lds_b128((gi32x4*)g, (si32x4*)l, 0, 0);
#else
    *(u32x4*)l = *(const u32x4*)g;
#endif
}
__device__ __forceinline__ void wait_async_done() {
#if HAVE_ASYNC_LDS
#if defined(__has_builtin) && __has_builtin(__builtin_amdgcn_s_wait_asynccnt)
    __builtin_amdgcn_s_wait_asynccnt(0);
#else
    asm volatile("s_wait_asynccnt 0" ::: "memory");
#endif
#endif
}

// ---------------- weight precompute: collapse rank-1 PHM into H^T (bf16) ----------------
__global__ __launch_bounds__(256) void phm_precompute_k(
    const float* __restrict__ phm_d, const float* __restrict__ wl_d, const float* __restrict__ wr_d,
    const float* __restrict__ phm_u, const float* __restrict__ wl_u, const float* __restrict__ wr_u,
    unsigned short* __restrict__ hd, unsigned short* __restrict__ hu) {
    int idx = blockIdx.x * 256 + threadIdx.x;
    if (idx < 768 * 192) {
        // Hd_t[n][k], n=c*48+qq (out col), k=a*192+pp (in row)
        int n = idx / 768, k = idx % 768;
        int a = k / 192, pp = k % 192;
        int c = n / 48,  qq = n % 48;
        float s = 0.f;
#pragma unroll
        for (int i = 0; i < 4; ++i)
            s += phm_d[i*16 + a*4 + c] * wl_d[i*192 + pp] * wr_d[i*48 + qq];
        hd[idx] = f2bf(s);
    } else if (idx < 2 * 768 * 192) {
        int j = idx - 768 * 192;
        // Hu_t[n][k], n<768: n=c*192+qq, k=a*48+pp
        int n = j / 192, k = j % 192;
        int a = k / 48,  pp = k % 48;
        int c = n / 192, qq = n % 192;
        float s = 0.f;
#pragma unroll
        for (int i = 0; i < 4; ++i)
            s += phm_u[i*16 + a*4 + c] * wl_u[i*48 + pp] * wr_u[i*192 + qq];
        hu[j] = f2bf(s);
    }
}

// ---------------- fused down-GEMM + gelu + up-GEMM ----------------
// 256 threads = 8 waves; 256 rows/block; wave w owns rows [32w, 32w+32) (two 16-row strips).
#define BROW 20   // LDS B-slice row stride in dwords (16 payload + 4 pad): 16B-aligned, conflict-free
__global__ __launch_bounds__(256) void phm_fused_k(
    const float* __restrict__ x,
    const unsigned int* __restrict__ hd_dw,   // Hd_t [192][768] bf16 as dwords (384/row)
    const unsigned int* __restrict__ hu_dw,   // Hu_t [768][192] bf16 as dwords (96/row)
    const float* __restrict__ bias_d,
    const float* __restrict__ bias_u,
    float* __restrict__ out) {
    __shared__ unsigned int Bsm[2][192 * BROW];   // 30720 B, double-buffered B slices
    __shared__ unsigned int Zs_u[256 * 100];      // 102400 B, z tile bf16 (row stride 100 dw)

    const int t    = threadIdx.x;
    const int lane = t & 31;
    const int wave = t >> 5;
    const int l15  = lane & 15;
    const int hi   = lane >> 4;
    const int blockRow = blockIdx.x * 256;
    const int rowBase  = wave * 32;                          // block-relative wave rows
    const float* xrow0 = x + (size_t)(blockRow + rowBase + l15) * 768;
    const float* xrow1 = xrow0 + (size_t)16 * 768;

    // ---- stage helpers (cooperative, whole block; each lane moves 16B chunks) ----
    auto stage1 = [&](int kc, int buf) {            // Hd_t slice: 192 rows x 16 dw
#pragma unroll
        for (int j = 0; j < 3; ++j) {
            int q = t + j * 256;                     // uint4 index, 768 total
            int n = q >> 2, kkd = (q & 3) * 4;
            ld_b128_to_lds(hd_dw + n * 384 + kc * 16 + kkd, &Bsm[buf][n * BROW + kkd]);
        }
    };
    auto stage2 = [&](int s, int buf) {             // Hu_t slice: 128 rows x 16 dw
        int g = s / 6, kc2 = s % 6;
#pragma unroll
        for (int j = 0; j < 2; ++j) {
            int q = t + j * 256;                     // uint4 index, 512 total
            int nl = q >> 2, kkd = (q & 3) * 4;
            ld_b128_to_lds(hu_dw + (size_t)(g * 128 + nl) * 96 + kc2 * 16 + kkd,
                           &Bsm[buf][nl * BROW + kkd]);
        }
    };
    auto makeA = [&](const float* xr, int kA, Frag& A) {
        f32x4 a0 = __builtin_nontemporal_load((const f32x4*)(xr + kA));
        f32x4 a1 = __builtin_nontemporal_load((const f32x4*)(xr + kA + 4));
        f32x4 a2 = __builtin_nontemporal_load((const f32x4*)(xr + kA + 16));
        f32x4 a3 = __builtin_nontemporal_load((const f32x4*)(xr + kA + 20));
        A.u[0] = pack2(a0.x, a0.y); A.u[1] = pack2(a0.z, a0.w);
        A.u[2] = pack2(a1.x, a1.y); A.u[3] = pack2(a1.z, a1.w);
        A.u[4] = pack2(a2.x, a2.y); A.u[5] = pack2(a2.z, a2.w);
        A.u[6] = pack2(a3.x, a3.y); A.u[7] = pack2(a3.z, a3.w);
    };
    auto loadB = [&](const unsigned int* base, int nt, Frag& B) {
        const unsigned int* bp = base + (nt * 16 + l15) * BROW + hi * 4;
        u32x4 b0 = *(const u32x4*)bp;
        u32x4 b1 = *(const u32x4*)(bp + 8);
        B.u[0] = b0.x; B.u[1] = b0.y; B.u[2] = b0.z; B.u[3] = b0.w;
        B.u[4] = b1.x; B.u[5] = b1.y; B.u[6] = b1.z; B.u[7] = b1.w;
    };

    // ================= phase 1: z = x @ Hd (32x192 per wave) =================
    v8f acc[24];                                     // [strip][nt] : strip*12 + nt
#pragma unroll
    for (int i = 0; i < 24; ++i) acc[i] = v8f{};

    stage1(0, 0);
    wait_async_done();
    __syncthreads();
#pragma unroll 1
    for (int kc = 0; kc < 24; ++kc) {
        int cur = kc & 1;
        if (kc + 1 < 24) stage1(kc + 1, cur ^ 1);
        // A fragments: lanes 0-15: K {0..7,16..23}, lanes 16-31: +8
        int kA = kc * 32 + hi * 8;
        Frag A0, A1;
        makeA(xrow0, kA, A0);
        makeA(xrow1, kA, A1);
#pragma unroll
        for (int nt = 0; nt < 12; ++nt) {
            Frag B;
            loadB(Bsm[cur], nt, B);                  // B reused by both strips
            acc[nt] = __builtin_amdgcn_wmma_f32_16x16x32_bf16(
                false, A0.v, false, B.v, (short)0, acc[nt], false, false);
            acc[12 + nt] = __builtin_amdgcn_wmma_f32_16x16x32_bf16(
                false, A1.v, false, B.v, (short)0, acc[12 + nt], false, false);
        }
        wait_async_done();   // next-buffer staging done (overlapped with WMMAs)
        __syncthreads();
    }

    // bias + gelu_new, spill z to LDS as bf16 (A-layout friendly, row stride 200 ushorts)
    unsigned short* Zs = (unsigned short*)Zs_u;
#pragma unroll
    for (int sp = 0; sp < 2; ++sp) {
#pragma unroll
        for (int nt = 0; nt < 12; ++nt) {
            int col = nt * 16 + l15;
            float b = bias_d[col];
#pragma unroll
            for (int r = 0; r < 8; ++r) {
                float v = gelu_new_f(acc[sp * 12 + nt][r] + b);
                Zs[(rowBase + sp * 16 + r + 8 * hi) * 200 + col] = f2bf(v);
            }
        }
    }
    __syncthreads();

    // ================= phase 2: out = z @ Hu (32x768 per wave, 6 N-groups) =================
    v8f acc2[16];                                    // [strip][nt] : strip*8 + nt
    stage2(0, 0);
    wait_async_done();
    __syncthreads();
#pragma unroll 1
    for (int s = 0; s < 36; ++s) {
        int cur = s & 1;
        if (s + 1 < 36) stage2(s + 1, cur ^ 1);
        int g = s / 6, kc2 = s % 6;
        if (kc2 == 0) {
#pragma unroll
            for (int i = 0; i < 16; ++i) acc2[i] = v8f{};
        }
        // A fragments from z in LDS
        Frag A0, A1;
        {
            const unsigned int* zp0 = Zs_u + (rowBase + l15) * 100 + kc2 * 16 + hi * 4;
            const unsigned int* zp1 = zp0 + 16 * 100;
            u32x4 za = *(const u32x4*)zp0, zb = *(const u32x4*)(zp0 + 8);
            A0.u[0] = za.x; A0.u[1] = za.y; A0.u[2] = za.z; A0.u[3] = za.w;
            A0.u[4] = zb.x; A0.u[5] = zb.y; A0.u[6] = zb.z; A0.u[7] = zb.w;
            u32x4 zc = *(const u32x4*)zp1, zd = *(const u32x4*)(zp1 + 8);
            A1.u[0] = zc.x; A1.u[1] = zc.y; A1.u[2] = zc.z; A1.u[3] = zc.w;
            A1.u[4] = zd.x; A1.u[5] = zd.y; A1.u[6] = zd.z; A1.u[7] = zd.w;
        }
#pragma unroll
        for (int nt = 0; nt < 8; ++nt) {
            Frag B;
            loadB(Bsm[cur], nt, B);                  // B reused by both strips
            acc2[nt] = __builtin_amdgcn_wmma_f32_16x16x32_bf16(
                false, A0.v, false, B.v, (short)0, acc2[nt], false, false);
            acc2[8 + nt] = __builtin_amdgcn_wmma_f32_16x16x32_bf16(
                false, A1.v, false, B.v, (short)0, acc2[8 + nt], false, false);
        }
        if (kc2 == 5) {
#pragma unroll
            for (int sp = 0; sp < 2; ++sp) {
#pragma unroll
                for (int nt = 0; nt < 8; ++nt) {
                    int col = g * 128 + nt * 16 + l15;
                    float bu = bias_u[col];
#pragma unroll
                    for (int r = 0; r < 8; ++r) {
                        size_t row = (size_t)(blockRow + rowBase + sp * 16 + r + 8 * hi);
                        __builtin_nontemporal_store(acc2[sp * 8 + nt][r] + bu,
                                                    out + row * 768 + col);
                    }
                }
            }
        }
        wait_async_done();
        __syncthreads();
    }
}

extern "C" void kernel_launch(void* const* d_in, const int* in_sizes, int n_in,
                              void* d_out, int out_size, void* d_ws, size_t ws_size,
                              hipStream_t stream) {
    const float* x      = (const float*)d_in[0];
    const float* phm_d  = (const float*)d_in[1];
    const float* wl_d   = (const float*)d_in[2];
    const float* wr_d   = (const float*)d_in[3];
    const float* bias_d = (const float*)d_in[4];
    const float* phm_u  = (const float*)d_in[5];
    const float* wl_u   = (const float*)d_in[6];
    const float* wr_u   = (const float*)d_in[7];
    const float* bias_u = (const float*)d_in[8];
    float* out = (float*)d_out;

    // workspace: Hd_t (768*192 bf16) then Hu_t (768*192 bf16) = 576 KB total, L2-resident
    unsigned short* hd = (unsigned short*)d_ws;
    unsigned short* hu = hd + 768 * 192;

    int M = in_sizes[0] / 768;   // 32768

    phm_precompute_k<<<(2 * 768 * 192 + 255) / 256, 256, 0, stream>>>(
        phm_d, wl_d, wr_d, phm_u, wl_u, wr_u, hd, hu);
    phm_fused_k<<<M / 256, 256, 0, stream>>>(
        x, (const unsigned int*)hd, (const unsigned int*)hu, bias_d, bias_u, out);
}
